// HouseholdODEFunc_49460843381260
// MI455X (gfx1250) — matrix operand order repaired and checked by reading, not compiled, not run on hardware
//
#include <hip/hip_runtime.h>
#include <hip/hip_bf16.h>

// ---------------------------------------------------------------------------
// MI455X (gfx1250) implementation of HouseholdODEFunc.
// Strategy: all big GEMMs run on v_wmma_f32_16x16x32_bf16 (fp32 accumulate),
// fp32 operands converted to bf16 while staging into LDS.  Leave-one-out
// interaction is algebraically folded into Weff = Wi_top - Wi_bot/(N-1)
// (combined during B staging) plus a per-layer constant bias vector d[l].
// ---------------------------------------------------------------------------

typedef __attribute__((ext_vector_type(16))) __bf16 v16bf;
typedef __attribute__((ext_vector_type(8)))  float  v8f;

__device__ __forceinline__ unsigned short bfbits(float f) {
    unsigned u = __float_as_uint(f);
    u += 0x7FFFu + ((u >> 16) & 1u);           // round-to-nearest-even
    return (unsigned short)(u >> 16);
}
__device__ __forceinline__ __bf16 f2bf(float f) {
    unsigned short s = bfbits(f);
    __bf16 r;
    __builtin_memcpy(&r, &s, 2);
    return r;
}

union Frag {            // one WMMA 16-lane bf16 fragment = 32 bytes
    v16bf v;
    uint4 q[2];
};

// ---------------------------------------------------------------------------
// Tiled WMMA GEMM:  D[M,Nn] = epi( A[M,K] @ (B1 + alpha*B2)[K,Nn] + bias )
//   EPI 0: relu(x + bias)
//   EPI 1: Cres + relu(x + bias)      (residual add, in-place safe)
//   EPI 2: 0.1f * (x + bias)
// Block tile 128x128, K-tile 64, 256 threads (8 wave32s, each 32x64).
// ---------------------------------------------------------------------------
template <int EPI>
__global__ __launch_bounds__(256) void gemm_wmma_kernel(
    const float* __restrict__ A, int lda,
    const float* __restrict__ B1, const float* __restrict__ B2, float alpha,
    int ldb,
    const float* __restrict__ bias,
    const float* __restrict__ Cres,
    float* __restrict__ D, int ldd,
    int M, int Nn, int K)
{
    constexpr int BM = 128, BN = 128, BK = 64;

    __shared__ __align__(16) __bf16 sA[BM * BK];   // [m][k]  16 KB
    __shared__ __align__(16) __bf16 sB[BN * BK];   // [n][k]  16 KB (transposed)

    const int tid  = threadIdx.x;
    const int lane = tid & 31;
    const int wid  = tid >> 5;          // 0..7
    const int wm   = wid & 3;           // wave row  -> 32 rows each
    const int wn   = wid >> 2;          // wave col  -> 64 cols each
    const int l15  = lane & 15;
    const int hf   = lane >> 4;         // 0 or 1 (lane half)

    const int m0 = blockIdx.y * BM;
    const int n0 = blockIdx.x * BN;

    v8f acc[2][4];
#pragma unroll
    for (int i = 0; i < 2; ++i)
#pragma unroll
        for (int j = 0; j < 4; ++j)
#pragma unroll
            for (int r = 0; r < 8; ++r) acc[i][j][r] = 0.0f;

    // staging decompositions
    const int ac4 = tid & 15;           // A: 16 float4 groups across 64 cols
    const int ar0 = tid >> 4;           // A: 16 rows per pass
    const int bc4 = tid & 31;           // B: 32 float4 groups across 128 cols
    const int br0 = tid >> 5;           // B: 8 k-rows per pass

    for (int k0 = 0; k0 < K; k0 += BK) {
        // ---- stage A tile: 128x64 fp32 -> bf16 row-major ----
#pragma unroll
        for (int p = 0; p < 8; ++p) {
            const int r = ar0 + p * 16;
            const float4 a4 =
                *(const float4*)(A + (size_t)(m0 + r) * lda + k0 + ac4 * 4);
            const unsigned lo = ((unsigned)bfbits(a4.y) << 16) | bfbits(a4.x);
            const unsigned hi = ((unsigned)bfbits(a4.w) << 16) | bfbits(a4.z);
            *reinterpret_cast<uint2*>(&sA[r * BK + ac4 * 4]) =
                make_uint2(lo, hi);
            if (k0 + BK < K)   // hint next K tile into cache
                __builtin_prefetch(A + (size_t)(m0 + r) * lda + k0 + BK +
                                       ac4 * 4, 0, 1);
        }
        // ---- stage B tile: 64x128 fp32 -> bf16 transposed [n][k] ----
#pragma unroll
        for (int p = 0; p < 8; ++p) {
            const int kr = br0 + p * 8;
            const float4 b4 =
                *(const float4*)(B1 + (size_t)(k0 + kr) * ldb + n0 + bc4 * 4);
            float v0 = b4.x, v1 = b4.y, v2 = b4.z, v3 = b4.w;
            if (B2) {
                const float4 c4 =
                    *(const float4*)(B2 + (size_t)(k0 + kr) * ldb + n0 + bc4 * 4);
                v0 += alpha * c4.x; v1 += alpha * c4.y;
                v2 += alpha * c4.z; v3 += alpha * c4.w;
            }
            const int nb = bc4 * 4;
            sB[(nb + 0) * BK + kr] = f2bf(v0);
            sB[(nb + 1) * BK + kr] = f2bf(v1);
            sB[(nb + 2) * BK + kr] = f2bf(v2);
            sB[(nb + 3) * BK + kr] = f2bf(v3);
        }
        __syncthreads();

        // ---- compute: 2 WMMA K-steps of 32 ----
#pragma unroll
        for (int kk = 0; kk < BK; kk += 32) {
            Frag af[2], bf[4];
#pragma unroll
            for (int sm = 0; sm < 2; ++sm) {
                const int base = (wm * 32 + sm * 16 + l15) * BK + kk + hf * 8;
                af[sm].q[0] = *(const uint4*)(sA + base);
                af[sm].q[1] = *(const uint4*)(sA + base + 16);
            }
#pragma unroll
            for (int sn = 0; sn < 4; ++sn) {
                const int base = (wn * 64 + sn * 16 + l15) * BK + kk + hf * 8;
                bf[sn].q[0] = *(const uint4*)(sB + base);
                bf[sn].q[1] = *(const uint4*)(sB + base + 16);
            }
#pragma unroll
            for (int sm = 0; sm < 2; ++sm)
#pragma unroll
                for (int sn = 0; sn < 4; ++sn)
                    acc[sm][sn] = __builtin_amdgcn_wmma_f32_16x16x32_bf16(
                        false, af[sm].v, false, bf[sn].v,
                        (short)0, acc[sm][sn], false, false);
        }
        __syncthreads();
    }

    // ---- epilogue ----
#pragma unroll
    for (int sm = 0; sm < 2; ++sm) {
        const int gm = m0 + wm * 32 + sm * 16 + hf * 8;
#pragma unroll
        for (int sn = 0; sn < 4; ++sn) {
            const int gn = n0 + wn * 64 + sn * 16 + l15;
            const float bs = bias[gn];
#pragma unroll
            for (int r = 0; r < 8; ++r) {
                const float x = acc[sm][sn][r] + bs;
                const size_t idx = (size_t)(gm + r) * ldd + gn;
                if (EPI == 0) {
                    D[idx] = fmaxf(x, 0.0f);
                } else if (EPI == 1) {
                    D[idx] = Cres[idx] + fmaxf(x, 0.0f);
                } else {
                    D[idx] = 0.1f * x;
                }
            }
        }
    }
}

// ---------------------------------------------------------------------------
// c[j] = b_in[j] + sum_k (t*Wt[k]+bt[k]) * Win[512+k][j]   (fold time embed)
// ---------------------------------------------------------------------------
__global__ __launch_bounds__(256) void timec_kernel(
    const float* __restrict__ t, const float* __restrict__ Wt,
    const float* __restrict__ bt, const float* __restrict__ Win,
    const float* __restrict__ b_in, float* __restrict__ c)
{
    const int j = blockIdx.x * 256 + threadIdx.x;   // 0..1023
    const float tv = t[0];
    float s = b_in[j];
    for (int k = 0; k < 128; ++k)
        s += (tv * Wt[k] + bt[k]) * Win[(size_t)(512 + k) * 1024 + j];
    c[j] = s;
}

// ---------------------------------------------------------------------------
// In-place LayerNorm over H=1024 features; one block per row.
// ---------------------------------------------------------------------------
__global__ __launch_bounds__(256) void ln_kernel(
    float* __restrict__ hi, const float* __restrict__ g,
    const float* __restrict__ b, int H)
{
    const int tid = threadIdx.x;
    float* p = hi + (size_t)blockIdx.x * H;
    float4 v = *(float4*)(p + tid * 4);

    __shared__ float red[256];
    red[tid] = v.x + v.y + v.z + v.w;
    __syncthreads();
    for (int o = 128; o > 0; o >>= 1) {
        if (tid < o) red[tid] += red[tid + o];
        __syncthreads();
    }
    const float mu = red[0] * (1.0f / 1024.0f);
    __syncthreads();

    const float dx = v.x - mu, dy = v.y - mu, dz = v.z - mu, dw = v.w - mu;
    red[tid] = dx * dx + dy * dy + dz * dz + dw * dw;
    __syncthreads();
    for (int o = 128; o > 0; o >>= 1) {
        if (tid < o) red[tid] += red[tid + o];
        __syncthreads();
    }
    const float rs = rsqrtf(red[0] * (1.0f / 1024.0f) + 1e-5f);

    const int j = tid * 4;
    v.x = dx * rs * g[j + 0] + b[j + 0];
    v.y = dy * rs * g[j + 1] + b[j + 1];
    v.z = dz * rs * g[j + 2] + b[j + 2];
    v.w = dw * rs * g[j + 3] + b[j + 3];
    *(float4*)(p + tid * 4) = v;
}

// colsum[c] = sum_r hi[r][c]
__global__ __launch_bounds__(256) void colsum_kernel(
    const float* __restrict__ hi, float* __restrict__ cs, int N, int H)
{
    const int col = blockIdx.x * 256 + threadIdx.x;
    float s = 0.0f;
    for (int r = 0; r < N; ++r) s += hi[(size_t)r * H + col];
    cs[col] = s;
}

// d[j] = bi[j] + (colsum/(N-1)) @ Wi_bot[:, j]
__global__ __launch_bounds__(256) void dvec_kernel(
    const float* __restrict__ cs, const float* __restrict__ WiBot,
    const float* __restrict__ bi, float* __restrict__ d, float inv, int H)
{
    const int j = blockIdx.x * 256 + threadIdx.x;
    float s = bi[j];
    for (int k = 0; k < H; ++k)
        s += cs[k] * inv * WiBot[(size_t)k * H + j];
    d[j] = s;
}

// ---------------------------------------------------------------------------
extern "C" void kernel_launch(void* const* d_in, const int* in_sizes, int n_in,
                              void* d_out, int out_size, void* d_ws,
                              size_t ws_size, hipStream_t stream)
{
    const float* t     = (const float*)d_in[0];
    const float* state = (const float*)d_in[1];
    const float* Wt    = (const float*)d_in[2];
    const float* bt    = (const float*)d_in[3];
    const float* Win   = (const float*)d_in[4];
    const float* b_in  = (const float*)d_in[5];
    const float* Wh    = (const float*)d_in[6];
    const float* bh    = (const float*)d_in[7];
    const float* Wi    = (const float*)d_in[8];
    const float* bi    = (const float*)d_in[9];
    const float* ln_g  = (const float*)d_in[10];
    const float* ln_b  = (const float*)d_in[11];
    const float* Wout  = (const float*)d_in[12];
    const float* bout  = (const float*)d_in[13];
    float* out = (float*)d_out;

    const int TOT = 640, H = 1024;
    const int N = in_sizes[1] / TOT;                 // 16384
    const float invNm1 = 1.0f / (float)(N - 1);

    float* h    = (float*)d_ws;                      // [N,H] fp32
    float* hi   = h  + (size_t)N * H;                // [N,H] fp32
    float* cvec = hi + (size_t)N * H;                // [H]
    float* csum = cvec + H;                          // [H]
    float* dv   = csum + H;                          // [H]

    const dim3 blk(256);
    const dim3 gH(H / 128, N / 128);                 // (8,128)
    const dim3 gO(TOT / 128, N / 128);               // (5,128)

    // fold t into first-layer bias; first GEMM shrinks to K=512
    timec_kernel<<<4, blk, 0, stream>>>(t, Wt, bt, Win, b_in, cvec);

    // h = relu(state[:, :512] @ Win[:512] + c)
    gemm_wmma_kernel<0><<<gH, blk, 0, stream>>>(
        state, TOT, Win, nullptr, 0.0f, H, cvec, nullptr, h, H, N, H, 512);

    for (int l = 0; l < 3; ++l) {
        const float* Whl   = Wh + (size_t)l * H * H;
        const float* WiTop = Wi + (size_t)l * 2 * H * H;
        const float* WiBot = WiTop + (size_t)H * H;

        // hi = relu(h @ Wh[l] + bh[l])
        gemm_wmma_kernel<0><<<gH, blk, 0, stream>>>(
            h, H, Whl, nullptr, 0.0f, H, bh + l * H, nullptr, hi, H, N, H, H);
        // hi = LayerNorm(hi)
        ln_kernel<<<N, blk, 0, stream>>>(hi, ln_g + l * H, ln_b + l * H, H);
        // colsum over rows, then d[l] = bi + (colsum/(N-1)) @ Wi_bot
        colsum_kernel<<<4, blk, 0, stream>>>(hi, csum, N, H);
        dvec_kernel<<<4, blk, 0, stream>>>(csum, WiBot, bi + l * H, dv,
                                           invNm1, H);
        // h = h + relu(hi @ (Wi_top - Wi_bot/(N-1)) + d[l])
        gemm_wmma_kernel<1><<<gH, blk, 0, stream>>>(
            hi, H, WiTop, WiBot, -invNm1, H, dv, h, h, H, N, H, H);
    }

    // out = 0.1 * (h @ Wout + bout)
    gemm_wmma_kernel<2><<<gO, blk, 0, stream>>>(
        h, H, Wout, nullptr, 0.0f, TOT, bout, nullptr, out, TOT, N, TOT, H);
}
